// TokenEmbedding_34935263985768
// MI455X (gfx1250) — matrix operand order repaired and checked by reading, not compiled, not run on hardware
//
#include <hip/hip_runtime.h>

typedef __attribute__((ext_vector_type(16))) _Float16 v16h;
typedef __attribute__((ext_vector_type(8)))  _Float16 v8h;
typedef __attribute__((ext_vector_type(8)))  float    v8f;

namespace {
constexpr int kTao    = 2;
constexpr int kM      = 3;
constexpr int kCin    = 8;
constexpr int kDmodel = 128;
constexpr int kFeat   = 32;          // kCin * (kM + 1)
constexpr int kS      = 8192;
constexpr int kKtot   = 128;         // 3*kFeat (conv taps) + kFeat (channel enc)
constexpr int kPitch  = 136;         // LDS row pitch in halves (272B, 16B aligned)
constexpr int kTokPerBlk = 64;       // 4 waves x 16 tokens
constexpr int kWin    = 72;          // x time-window: [tok_base-7, tok_base+64]
}

// ---------------------------------------------------------------------------
// Kernel 1: fuse conv (width-3) + projection into a single 128x128 f16 matrix
//   Wt[n][s*32+i] = sum_j proj_w[n][j] * conv_w[j][i][s]   (s = 0..2)
//   Wt[n][96+i]   = proj_w[n][128+i]
//   bias[n]       = proj_b[n] + sum_j proj_w[n][j] * conv_b[j]
// ---------------------------------------------------------------------------
__global__ __launch_bounds__(128) void fuse_weights_kernel(
    const float* __restrict__ conv_w,   // [128][32][3]
    const float* __restrict__ conv_b,   // [128]
    const float* __restrict__ proj_w,   // [128][160]
    const float* __restrict__ proj_b,   // [128]
    _Float16* __restrict__ wt,          // [128][128] f16, k contiguous
    float* __restrict__ bias)           // [128]
{
    __shared__ float prow[kDmodel + kFeat];
    const int n = blockIdx.x;
    const int k = threadIdx.x;
    prow[k] = proj_w[n * 160 + k];
    if (k < kFeat) prow[kDmodel + k] = proj_w[n * 160 + kDmodel + k];
    __syncthreads();

    float v;
    if (k < 3 * kFeat) {
        const int s = k >> 5;        // conv tap
        const int i = k & 31;        // input feature
        float acc = 0.f;
#pragma unroll 4
        for (int j = 0; j < kDmodel; ++j)
            acc += prow[j] * conv_w[j * (kFeat * 3) + i * 3 + s];
        v = acc;
    } else {
        v = prow[kDmodel + (k - 3 * kFeat)];
    }
    wt[n * kKtot + k] = (_Float16)v;

    if (k == 0) {
        float bb = proj_b[n];
        for (int j = 0; j < kDmodel; ++j) bb += prow[j] * conv_b[j];
        bias[n] = bb;
    }
}

// ---------------------------------------------------------------------------
// Kernel 2: fused gather + GEMM via V_WMMA_F32_16X16X32_F16.
// 128 threads (4 waves) per block; 64 consecutive tokens of one batch row.
// Accumulators zero-init (inline SRC2=0); bias added in the store loop.
// ---------------------------------------------------------------------------
__global__ __launch_bounds__(128) void fused_token_kernel(
    const float* __restrict__ x,        // [64][8192][8]
    const float* __restrict__ ce,       // [64][8192][32]
    const _Float16* __restrict__ wt,    // [128][128]
    const float* __restrict__ bias,     // [128]
    float* __restrict__ out)            // [64][8192][128]
{
    __shared__ _Float16 sw[kDmodel * kPitch];     // fused weights, k-major rows
    __shared__ _Float16 sv[kTokPerBlk * kPitch];  // per-token feature vectors
    __shared__ float    sx[kWin * kCin];          // x time window (mod-S times)
    __shared__ float    sbias[kDmodel];

    const int tid = threadIdx.x;
    const int gt0 = blockIdx.x * kTokPerBlk;
    const int b = gt0 >> 13;            // / kS
    const int tok_base = gt0 & (kS - 1);

    const size_t xrow  = (size_t)b * kS * kCin;
    const size_t cerow = (size_t)b * kS * kFeat;

    // ---- Phase 1: stage weights, bias, and the x time-window --------------
    for (int idx = tid; idx < kDmodel * (kKtot / 8); idx += blockDim.x) {
        const int n = idx >> 4;         // row
        const int c = idx & 15;         // 8-half chunk within row
        *reinterpret_cast<uint4*>(&sw[n * kPitch + c * 8]) =
            *reinterpret_cast<const uint4*>(wt + n * kKtot + c * 8);
    }
    if (tid < kDmodel) sbias[tid] = bias[tid];

    // x window: position i holds time (tok_base - 7 + i) mod S.
    // Using mod-S times here makes the circular 'wrap' padding fall out
    // naturally when feats are indexed by *unwrapped* positions below.
    for (int idx = tid; idx < kWin * kCin; idx += blockDim.x) {
        const int i = idx >> 3;
        const int c = idx & 7;
        const int w = (tok_base - 7 + i + kS) & (kS - 1);
        sx[idx] = x[xrow + (size_t)w * kCin + c];
    }
    __syncthreads();

    // ---- Phase 2: build per-token feature vectors (f16) -------------------
    //   sv[tl][0:96)  = [feats[t-1]; feats[t]; feats[t+1]]
    //   sv[tl][96:128)= ce[t]
    //   feats[tt][c*4+m] = (wrap(tt) >= 6) ? x[b, tt-2m, c] : 0
    // Quad-packed: 4 lags (m=0..3) of one channel -> one ds_store_b64.
    for (int idx = tid; idx < kTokPerBlk * 32; idx += blockDim.x) {
        const int q = idx & 31;
        if (q >= 24) continue;          // 24 quads = 96 halves per token
        const int tl = idx >> 5;
        const int s = q >> 3;           // time tap: t-1, t, t+1
        const int c = q & 7;            // channel
        const int p = tl + s + 6;       // window position of tt (unwrapped)
        int ttw = tok_base + tl + s - 1;
        if (ttw < 0)   ttw += kS;
        if (ttw >= kS) ttw -= kS;
        union { _Float16 h[4]; uint2 u; } pk;
        if (ttw >= kM * kTao) {
            pk.h[0] = (_Float16)sx[(p    ) * kCin + c];
            pk.h[1] = (_Float16)sx[(p - 2) * kCin + c];
            pk.h[2] = (_Float16)sx[(p - 4) * kCin + c];
            pk.h[3] = (_Float16)sx[(p - 6) * kCin + c];
        } else {
            pk.h[0] = pk.h[1] = pk.h[2] = pk.h[3] = (_Float16)0.f;
        }
        *reinterpret_cast<uint2*>(&sv[tl * kPitch + s * 32 + c * 4]) = pk.u;
    }
    // channel_encoding: float4 x2 -> 8 packed halves -> one ds_store_b128.
    for (int idx = tid; idx < kTokPerBlk * 4; idx += blockDim.x) {
        const int tl = idx >> 2;
        const int g  = idx & 3;
        const float4* src = reinterpret_cast<const float4*>(
            ce + cerow + (size_t)(tok_base + tl) * kFeat + g * 8);
        const float4 f0 = src[0];
        const float4 f1 = src[1];
        union { _Float16 h[8]; uint4 u; } pk;
        pk.h[0] = (_Float16)f0.x; pk.h[1] = (_Float16)f0.y;
        pk.h[2] = (_Float16)f0.z; pk.h[3] = (_Float16)f0.w;
        pk.h[4] = (_Float16)f1.x; pk.h[5] = (_Float16)f1.y;
        pk.h[6] = (_Float16)f1.z; pk.h[7] = (_Float16)f1.w;
        *reinterpret_cast<uint4*>(&sv[tl * kPitch + 96 + g * 8]) = pk.u;
    }
    __syncthreads();

    // ---- Phase 3: WMMA GEMM ----------------------------------------------
    const int lane  = tid & 31;
    const int wavei = tid >> 5;
    const int r     = lane & 15;
    const int hi    = lane >> 4;        // half-wave select
    const int row0  = wavei * 16;       // this wave's 16 tokens

    // A fragments (16 tokens x 32 K), 4 K-chunks.
    // Lane layout: row = lane%16; halves [hi*8, +8) and [hi*8+16, +8).
    v16h a[4];
#pragma unroll
    for (int kc = 0; kc < 4; ++kc) {
        const _Float16* p = &sv[(row0 + r) * kPitch + kc * 32 + hi * 8];
        v8h lo = *reinterpret_cast<const v8h*>(p);
        v8h hh = *reinterpret_cast<const v8h*>(p + 16);
        a[kc] = __builtin_shufflevector(lo, hh,
                 0,1,2,3,4,5,6,7,8,9,10,11,12,13,14,15);
    }

    const size_t outrow =
        ((size_t)b * kS + tok_base + row0 + hi * 8) * kDmodel;

#pragma unroll
    for (int nt = 0; nt < 8; ++nt) {
        // Zero-init: first WMMA uses inline SRC2 = 0 (no init instructions).
        v8f acc = {0.f, 0.f, 0.f, 0.f, 0.f, 0.f, 0.f, 0.f};
#pragma unroll
        for (int kc = 0; kc < 4; ++kc) {
            // B fragment (32 K x 16 N): lane holds col = lane%16,
            // K halves [hi*8, +8) and [hi*8+16, +8).
            const _Float16* p = &sw[(nt * 16 + r) * kPitch + kc * 32 + hi * 8];
            v8h lo = *reinterpret_cast<const v8h*>(p);
            v8h hh = *reinterpret_cast<const v8h*>(p + 16);
            v16h bfrag = __builtin_shufflevector(lo, hh,
                          0,1,2,3,4,5,6,7,8,9,10,11,12,13,14,15);
            acc = __builtin_amdgcn_wmma_f32_16x16x32_f16(
                false, a[kc], false, bfrag, (short)0, acc, false, false);
        }
        const float bc = sbias[nt * 16 + r];
#pragma unroll
        for (int i = 0; i < 8; ++i) {
            // C/D layout: VGPR i -> token row (i + hi*8), col = lane%16
            out[outrow + (size_t)i * kDmodel + nt * 16 + r] = acc[i] + bc;
        }
    }
}

// ---------------------------------------------------------------------------
extern "C" void kernel_launch(void* const* d_in, const int* in_sizes, int n_in,
                              void* d_out, int out_size, void* d_ws, size_t ws_size,
                              hipStream_t stream) {
    const float* x       = (const float*)d_in[0];
    const float* ce      = (const float*)d_in[1];
    const float* conv_w  = (const float*)d_in[2];
    const float* conv_b  = (const float*)d_in[3];
    const float* proj_w  = (const float*)d_in[4];
    const float* proj_b  = (const float*)d_in[5];
    float* out = (float*)d_out;

    _Float16* wt  = (_Float16*)d_ws;                                  // 32 KB
    float*    bia = (float*)((char*)d_ws + kDmodel * kKtot * sizeof(_Float16));

    fuse_weights_kernel<<<kDmodel, 128, 0, stream>>>(conv_w, conv_b,
                                                     proj_w, proj_b, wt, bia);

    const int total_tokens = in_sizes[0] / kCin;   // B * S = 524288
    fused_token_kernel<<<total_tokens / kTokPerBlk, 128, 0, stream>>>(
        x, ce, wt, bia, out);
}
